// MOE_Hard_Gate_72567767433424
// MI455X (gfx1250) — matrix operand order repaired and checked by reading, not compiled, not run on hardware
//
#include <hip/hip_runtime.h>
#include <hip/hip_bf16.h>

typedef __attribute__((ext_vector_type(16))) __bf16  v16bf;
typedef __attribute__((ext_vector_type(8)))  float   v8f;
typedef __attribute__((ext_vector_type(4)))  float   v4f;
typedef __attribute__((ext_vector_type(4)))  unsigned int v4u;
typedef __attribute__((ext_vector_type(2)))  unsigned int v2u;

union Frag16 { v16bf v; v4u q[2]; };

#define BLOCK   320
#define WAVES   10
#define ROWS    64       // rows (M) per block
#define MT      4        // 4 M-tiles of 16
#define KSTEPS  16       // 512 / 32
#define NTILES  20       // 320 / 16
#define LDA     520      // bf16 stride for staged x (512 + 8 pad)
#define LDH     321      // f32 stride for hidden (320 + 1 pad)
#define NHID    320
#define SMEM_BYTES (ROWS * LDH * 4)   // 82176 (> 64*520*2 = 66560 staging)
#define WFRAG_BYTES 327680            // 512*320*2

__device__ __forceinline__ unsigned short f2bf(float f) {
    union { float f; unsigned int u; } cv; cv.f = f;
    unsigned int u = cv.u;
    u += 0x7FFFu + ((u >> 16) & 1u);   // round-to-nearest-even
    return (unsigned short)(u >> 16);
}

// ---------------------------------------------------------------------------
// Prep: swizzle W1 = [gw1|aw1|bw1] (512 x 320 fp32) into per-lane WMMA
// B-fragment order (bf16) + concatenate layer-1 biases.
// Fragment address: ((t*16 + kk)*32 + lane)*16 + j  (16 bf16 per lane).
// B order used here (contiguous k, same as A): N = lane%16; lanes 0-15 hold K=K0+0..15 (2/VGPR),
// lanes 16-31 hold K=K0+16..31.
// ---------------------------------------------------------------------------
__global__ void moe_prep(const float* __restrict__ gw1, const float* __restrict__ aw1,
                         const float* __restrict__ bw1, const float* __restrict__ gb1,
                         const float* __restrict__ ab1, const float* __restrict__ bb1,
                         unsigned short* __restrict__ wfrag, float* __restrict__ bias1) {
    int gid = blockIdx.x * blockDim.x + threadIdx.x;
    if (gid < 512 * NHID) {
        int j    = gid & 15;
        int lane = (gid >> 4) & 31;
        int kk   = (gid >> 9) & 15;
        int t    = gid >> 13;
        int n    = t * 16 + (lane & 15);
        int k    = kk * 32 + j + ((lane >> 4) << 4);
        float wv;
        if (n < 64)       wv = gw1[k * 64  + n];
        else if (n < 192) wv = aw1[k * 128 + (n - 64)];
        else              wv = bw1[k * 128 + (n - 192)];
        wfrag[gid] = f2bf(wv);
    }
    if (gid < NHID) {
        bias1[gid] = (gid < 64) ? gb1[gid]
                   : (gid < 192) ? ab1[gid - 64]
                   : bb1[gid - 192];
    }
}

// ---------------------------------------------------------------------------
// Fused: stage x->bf16 LDS, layer-1 GEMM via v_wmma_f32_16x16x32_bf16,
// bias+ReLU -> LDS, layer-2 GEMVs + hard gate + masked store.
// ---------------------------------------------------------------------------
__global__ __launch_bounds__(BLOCK, 1)
void moe_main(const float* __restrict__ x, const unsigned short* __restrict__ wfrag,
              const float* __restrict__ bias1,
              const float* __restrict__ gw2, const float* __restrict__ gb2,
              const float* __restrict__ aw2, const float* __restrict__ ab2,
              const float* __restrict__ bw2, const float* __restrict__ bb2,
              float* __restrict__ out, int Brows) {
    extern __shared__ char smem[];
    unsigned short* sx = (unsigned short*)smem;   // staged bf16 x: [64][LDA]
    float*          sh = (float*)smem;            // hidden fp32:   [64][LDH] (reused)

    const int tid     = threadIdx.x;
    const int rowBase = blockIdx.x * ROWS;

    // ---- stage 64 rows of xf (concat halves) as bf16 into LDS -------------
    for (int e4 = tid; e4 < ROWS * 512 / 4; e4 += BLOCK) {
        int row = e4 >> 7;                 // /128 float4s per row
        int col = (e4 & 127) << 2;
        const float* src = (col < 256)
            ? (x + (size_t)(rowBase + row) * 256 + col)
            : (x + (size_t)(rowBase + row + Brows) * 256 + (col - 256));
        v4f f = *(const v4f*)src;
        unsigned int lo = (unsigned int)f2bf(f.x) | ((unsigned int)f2bf(f.y) << 16);
        unsigned int hi = (unsigned int)f2bf(f.z) | ((unsigned int)f2bf(f.w) << 16);
        v2u p; p.x = lo; p.y = hi;
        *(v2u*)(sx + row * LDA + col) = p;
    }
    __syncthreads();

    const int wave  = tid >> 5;
    const int lane  = tid & 31;
    const int lhalf = lane >> 4;       // 0 | 1
    const int lmod  = lane & 15;

    v8f acc[MT][2];
    #pragma unroll
    for (int m = 0; m < MT; ++m) { acc[m][0] = {}; acc[m][1] = {}; }

    // weight fragment base: tile t=2*wave and 2*wave+1, 32B per lane per frag
    const char* wb0 = (const char*)wfrag + (size_t)(2 * wave) * 16 * 1024 + lane * 32;
    const char* wb1 = wb0 + 16 * 1024;

    for (int kk = 0; kk < KSTEPS; ++kk) {
        Frag16 b0, b1;
        b0.q[0] = *(const v4u*)(wb0 + kk * 1024);
        b0.q[1] = *(const v4u*)(wb0 + kk * 1024 + 16);
        b1.q[0] = *(const v4u*)(wb1 + kk * 1024);
        b1.q[1] = *(const v4u*)(wb1 + kk * 1024 + 16);
        #pragma unroll
        for (int m = 0; m < MT; ++m) {
            Frag16 a;
            const unsigned short* ap = sx + (m * 16 + lmod) * LDA + kk * 32 + lhalf * 16;
            a.q[0] = *(const v4u*)(ap);        // K0 + {0..7 | 16..23}
            a.q[1] = *(const v4u*)(ap + 8);    // K0 + {8..15 | 24..31}
            acc[m][0] = __builtin_amdgcn_wmma_f32_16x16x32_bf16(
                false, a.v, false, b0.v, (short)0, acc[m][0], false, false);
            acc[m][1] = __builtin_amdgcn_wmma_f32_16x16x32_bf16(
                false, a.v, false, b1.v, (short)0, acc[m][1], false, false);
        }
    }
    __syncthreads();   // all waves done reading staged x before overwrite

    // ---- bias + ReLU, scatter hidden into LDS -----------------------------
    #pragma unroll
    for (int nt = 0; nt < 2; ++nt) {
        int coln = (2 * wave + nt) * 16 + lmod;
        float bv = bias1[coln];
        #pragma unroll
        for (int m = 0; m < MT; ++m) {
            #pragma unroll
            for (int v = 0; v < 8; ++v) {
                float h = acc[m][nt][v] + bv;
                h = h > 0.0f ? h : 0.0f;
                sh[(m * 16 + v + lhalf * 8) * LDH + coln] = h;
            }
        }
    }
    __syncthreads();

    // ---- layer 2: gate argmax + experts + masked store --------------------
    if (tid < ROWS) {
        const float* hr = sh + tid * LDH;
        float g0 = gb2[0], g1 = gb2[1];
        #pragma unroll 8
        for (int c = 0; c < 64; ++c) {
            float h = hr[c];
            g0 += h * gw2[2 * c]; g1 += h * gw2[2 * c + 1];
        }
        float a0 = ab2[0], a1 = ab2[1];
        #pragma unroll 8
        for (int c = 0; c < 128; ++c) {
            float h = hr[64 + c];
            a0 += h * aw2[2 * c]; a1 += h * aw2[2 * c + 1];
        }
        float b0 = bb2[0], b1 = bb2[1];
        #pragma unroll 8
        for (int c = 0; c < 128; ++c) {
            float h = hr[192 + c];
            b0 += h * bw2[2 * c]; b1 += h * bw2[2 * c + 1];
        }
        bool m_a = !(g1 > g0);   // jnp.argmax: first max wins on ties
        v4f o;
        o.x = m_a ? a0 : 0.0f;
        o.y = m_a ? a1 : 0.0f;
        o.z = m_a ? 0.0f : b0;
        o.w = m_a ? 0.0f : b1;
        *(v4f*)(out + (size_t)(rowBase + tid) * 4) = o;
    }
}

extern "C" void kernel_launch(void* const* d_in, const int* in_sizes, int n_in,
                              void* d_out, int out_size, void* d_ws, size_t ws_size,
                              hipStream_t stream) {
    const float* x   = (const float*)d_in[0];
    const float* gw1 = (const float*)d_in[1];
    const float* gb1 = (const float*)d_in[2];
    const float* gw2 = (const float*)d_in[3];
    const float* gb2 = (const float*)d_in[4];
    const float* aw1 = (const float*)d_in[5];
    const float* ab1 = (const float*)d_in[6];
    const float* aw2 = (const float*)d_in[7];
    const float* ab2 = (const float*)d_in[8];
    const float* bw1 = (const float*)d_in[9];
    const float* bb1 = (const float*)d_in[10];
    const float* bw2 = (const float*)d_in[11];
    const float* bb2 = (const float*)d_in[12];

    const int N2    = in_sizes[0] / 256;   // raw batch
    const int Brows = N2 / 2;              // 262144 gated rows

    unsigned short* wfrag = (unsigned short*)d_ws;
    float* bias1 = (float*)((char*)d_ws + WFRAG_BYTES);

    moe_prep<<<dim3((512 * NHID + 255) / 256), dim3(256), 0, stream>>>(
        gw1, aw1, bw1, gb1, ab1, bb1, wfrag, bias1);

    moe_main<<<dim3(Brows / ROWS), dim3(BLOCK), SMEM_BYTES, stream>>>(
        x, wfrag, bias1, gw2, gb2, aw2, ab2, bw2, bb2, (float*)d_out, Brows);
}
